// Original_LSTM_grid_layer_18030272708615
// MI455X (gfx1250) — compile-verified, matrix operand order checked by reading, not run on hardware
//
#include <hip/hip_runtime.h>
#include <stdint.h>

// ---------------------------------------------------------------------------
// Grid-LSTM persistent kernel for MI455X (gfx1250, wave32, WMMA bf16).
//   D=4 dofs, B=512 batch, T=64 steps, d=256.
//   - batch partitioned across 32 persistent workgroups (16 rows each);
//     recurrence never couples batch rows -> no inter-block sync needed.
//   - h,m state live in LDS (fp32) for all 64 steps; h also mirrored as bf16
//     for use as the WMMA A operand.
//   - weights packed once per launch to bf16, transposed [N][K] so B-fragments
//     load with the same two-b128-chunks-per-lane pattern as A-fragments.
// ---------------------------------------------------------------------------

typedef __attribute__((ext_vector_type(16))) __bf16 v16bf;
typedef __attribute__((ext_vector_type(8)))  float  v8f;
typedef __attribute__((ext_vector_type(8)))  __bf16 bf16x8;

constexpr int ND   = 4;         // dofs
constexpr int NB   = 512;       // batch
constexpr int NT   = 64;        // timesteps
constexpr int NDIM = 256;       // per-dof state dim
constexpr int ROWS = 16;        // batch rows per workgroup (= WMMA M tile)
constexpr int WG   = 512;       // 16 waves per workgroup
constexpr int KC   = 2 * NDIM;  // 512  : cell-stage K  (concat x,h)
constexpr int KG   = ND * NDIM; // 1024 : grid-stage K  (concat h over dofs)

__device__ __forceinline__ unsigned short f32_to_bf16(float f) {
  union { float f; uint32_t u; } v; v.f = f;
  uint32_t u = v.u;
  u += 0x7FFFu + ((u >> 16) & 1u);   // round to nearest even
  return (unsigned short)(u >> 16);
}

__device__ __forceinline__ float sigmoid_f(float x) { return 1.0f / (1.0f + __expf(-x)); }
__device__ __forceinline__ float tanh_f(float x) {
  float e = __expf(2.0f * x);
  return 1.0f - 2.0f / (e + 1.0f);
}

// 16x32 bf16 fragment: row-major source, row stride in elements.
// ISA 16-bit A layout: lane L (L&15 = row), k0 = 8*(L>=16); VGPR0-3 hold
// K=k0..k0+7, VGPR4-7 hold K=k0+16..k0+23  -> two contiguous 16B chunks.
__device__ __forceinline__ v16bf load_frag(const unsigned short* base, int stride, int lane) {
  const int r  = lane & 15;
  const int k0 = (lane >> 4) * 8;
  const unsigned short* p = base + r * stride + k0;
  bf16x8 lo = *reinterpret_cast<const bf16x8*>(p);
  bf16x8 hi = *reinterpret_cast<const bf16x8*>(p + 16);
  v16bf out;
#pragma unroll
  for (int i = 0; i < 8; ++i) { out[i] = lo[i]; out[i + 8] = hi[i]; }
  return out;
}

__device__ __forceinline__ v8f wmma_bf16(v16bf a, v16bf b, v8f c) {
  // D(f32 16x16) = A(16x32 bf16) * B(32x16 bf16) + C
  return __builtin_amdgcn_wmma_f32_16x16x32_bf16(false, a, false, b, (short)0, c, false, false);
}

// ---------------------------------------------------------------------------
// One-shot weight pack: fp32 [D][K][N] -> bf16 [gate][D][N][K] (transposed)
// cellW: 4 x [4][256][512]   gridW: 4 x [4][256][1024]
// ---------------------------------------------------------------------------
extern "C" __global__ void __launch_bounds__(256)
pack_weights(const float* __restrict__ wu,  const float* __restrict__ wf,
             const float* __restrict__ wo,  const float* __restrict__ wc,
             const float* __restrict__ gwu, const float* __restrict__ gwf,
             const float* __restrict__ gwo, const float* __restrict__ gwc,
             unsigned short* __restrict__ cellW, unsigned short* __restrict__ gridW) {
  const float* cw[4] = { wu, wf, wo, wc };
  const float* gw[4] = { gwu, gwf, gwo, gwc };
  const int cellTotal = 4 * ND * NDIM * KC;   // 2^21
  const int gridTotal = 4 * ND * NDIM * KG;   // 2^22
  for (int i = blockIdx.x * blockDim.x + threadIdx.x; i < cellTotal + gridTotal;
       i += gridDim.x * blockDim.x) {
    if (i < cellTotal) {
      int k  = i & (KC - 1);
      int n  = (i >> 9)  & (NDIM - 1);
      int dd = (i >> 17) & 3;
      int g  = i >> 19;
      cellW[i] = f32_to_bf16(cw[g][((size_t)dd * KC + k) * NDIM + n]);
    } else {
      int j  = i - cellTotal;
      int k  = j & (KG - 1);
      int n  = (j >> 10) & (NDIM - 1);
      int dd = (j >> 18) & 3;
      int g  = j >> 20;
      gridW[j] = f32_to_bf16(gw[g][((size_t)dd * KG + k) * NDIM + n]);
    }
  }
}

// ---------------------------------------------------------------------------
// Persistent grid-LSTM kernel. grid = 32 blocks x 512 threads (16 waves).
// Wave w -> (dof = w>>2, 64-column N-chunk = w&3); owns 4 gates x 4 N-tiles.
// ---------------------------------------------------------------------------
extern "C" __global__ void __launch_bounds__(WG, 1)
grid_lstm_persistent(const float* __restrict__ x,
                     const float* __restrict__ bu,  const float* __restrict__ bfv,
                     const float* __restrict__ bo,  const float* __restrict__ bc,
                     const float* __restrict__ gbu, const float* __restrict__ gbf,
                     const float* __restrict__ gbo, const float* __restrict__ gbc,
                     const unsigned short* __restrict__ cellW,
                     const unsigned short* __restrict__ gridW,
                     float* __restrict__ out) {
  extern __shared__ char smem[];
  unsigned short* xbuf = reinterpret_cast<unsigned short*>(smem);  // [ND][ROWS][NDIM] bf16
  unsigned short* hg   = xbuf + ND * ROWS * NDIM;                  // [ROWS][KG] bf16 (h concat)
  float* mbuf = reinterpret_cast<float*>(hg + ROWS * KG);          // [ND][ROWS][NDIM] fp32
  float* hbuf = mbuf + ND * ROWS * NDIM;                           // [ND][ROWS][NDIM] fp32

  const int tid   = threadIdx.x;
  const int wave  = tid >> 5;
  const int lane  = tid & 31;
  const int b0    = blockIdx.x * ROWS;
  const int dof   = wave >> 2;
  const int ncol0 = (wave & 3) * 64;
  const int rlo   = (lane >> 4) * 8;   // C-tile: lanes 16-31 hold M+8
  const int cl    = lane & 15;

  const float* biasC[4] = { bu,  bfv, bo,  bc  };
  const float* biasG[4] = { gbu, gbf, gbo, gbc };

  for (int i = tid; i < ND * ROWS * NDIM; i += WG) { mbuf[i] = 0.f; hbuf[i] = 0.f; }
  for (int i = tid; i < ROWS * KG; i += WG) hg[i] = 0;
  __syncthreads();

  v8f acc[16];
  const v8f zf = {};

  for (int t = 0; t < NT; ++t) {
    // ---- stage x_t tile into LDS as bf16 ----------------------------------
    for (int i = tid; i < ND * ROWS * NDIM; i += WG) {
      int dd = i >> 12, r = (i >> 8) & (ROWS - 1), j = i & (NDIM - 1);
      xbuf[i] = f32_to_bf16(x[(((size_t)dd * NB + (b0 + r)) * NT + t) * NDIM + j]);
    }
    __syncthreads();

    // ---- cell stage: [16 x 512] x [512 x 256] per (dof,gate) --------------
#pragma unroll
    for (int i = 0; i < 16; ++i) acc[i] = zf;
    for (int ks = 0; ks < KC / 32; ++ks) {
      const unsigned short* ab;
      int astr;
      if (ks < 8) { ab = xbuf + dof * ROWS * NDIM + ks * 32;      astr = NDIM; } // x half
      else        { ab = hg + dof * NDIM + (ks - 8) * 32;         astr = KG;   } // h_prev half
      v16bf a = load_frag(ab, astr, lane);
#pragma unroll
      for (int g = 0; g < 4; ++g)
#pragma unroll
        for (int nt = 0; nt < 4; ++nt) {
          const unsigned short* wb =
              cellW + ((size_t)((g * ND + dof) * NDIM + ncol0 + nt * 16)) * KC + ks * 32;
          acc[g * 4 + nt] = wmma_bf16(a, load_frag(wb, KC, lane), acc[g * 4 + nt]);
        }
    }
    __syncthreads();  // all waves finished reading h_prev from hg

#pragma unroll
    for (int nt = 0; nt < 4; ++nt) {
      const int col = ncol0 + nt * 16 + cl;
      const float bU = biasC[0][dof * NDIM + col];
      const float bF = biasC[1][dof * NDIM + col];
      const float bO = biasC[2][dof * NDIM + col];
      const float bC = biasC[3][dof * NDIM + col];
#pragma unroll
      for (int r = 0; r < 8; ++r) {
        const int row = rlo + r;
        const int idx = (dof * ROWS + row) * NDIM + col;
        float gu = sigmoid_f(acc[0 * 4 + nt][r] + bU);
        float gf = sigmoid_f(acc[1 * 4 + nt][r] + bF);
        float go = sigmoid_f(acc[2 * 4 + nt][r] + bO);
        float gc = tanh_f   (acc[3 * 4 + nt][r] + bC);
        float m  = gf * mbuf[idx] + gu * gc;
        float h  = tanh_f(go * m);
        mbuf[idx] = m;
        hbuf[idx] = h;
        hg[row * KG + dof * NDIM + col] = f32_to_bf16(h);
      }
    }
    __syncthreads();  // hg now holds cell-updated h for all dofs

    // ---- grid coupling stage: [16 x 1024] x [1024 x 256] ------------------
#pragma unroll
    for (int i = 0; i < 16; ++i) acc[i] = zf;
    for (int ks = 0; ks < KG / 32; ++ks) {
      v16bf a = load_frag(hg + ks * 32, KG, lane);
#pragma unroll
      for (int g = 0; g < 4; ++g)
#pragma unroll
        for (int nt = 0; nt < 4; ++nt) {
          const unsigned short* wb =
              gridW + ((size_t)((g * ND + dof) * NDIM + ncol0 + nt * 16)) * KG + ks * 32;
          acc[g * 4 + nt] = wmma_bf16(a, load_frag(wb, KG, lane), acc[g * 4 + nt]);
        }
    }
    __syncthreads();  // all waves finished reading hg

#pragma unroll
    for (int nt = 0; nt < 4; ++nt) {
      const int col = ncol0 + nt * 16 + cl;
      const float bU = biasG[0][dof * NDIM + col];
      const float bF = biasG[1][dof * NDIM + col];
      const float bO = biasG[2][dof * NDIM + col];
      const float bC = biasG[3][dof * NDIM + col];
#pragma unroll
      for (int r = 0; r < 8; ++r) {
        const int row = rlo + r;
        const int idx = (dof * ROWS + row) * NDIM + col;
        float gu = sigmoid_f(acc[0 * 4 + nt][r] + bU);
        float gf = sigmoid_f(acc[1 * 4 + nt][r] + bF);
        float go = sigmoid_f(acc[2 * 4 + nt][r] + bO);
        float gc = tanh_f   (acc[3 * 4 + nt][r] + bC);
        float m  = gf * mbuf[idx] + gu * gc;
        float h  = tanh_f(go * m);
        mbuf[idx] = m;
        hbuf[idx] = h;
        hg[row * KG + dof * NDIM + col] = f32_to_bf16(h);
      }
    }
    __syncthreads();  // next timestep reads updated hg
  }

  // ---- emit outputs: out = stack([h, m]) each [ND][NB][NDIM] fp32 ---------
  for (int i = tid; i < ND * ROWS * NDIM; i += WG) {
    int dd = i >> 12, r = (i >> 8) & (ROWS - 1), j = i & (NDIM - 1);
    size_t o = ((size_t)dd * NB + (b0 + r)) * NDIM + j;
    out[o] = hbuf[i];
    out[(size_t)ND * NB * NDIM + o] = mbuf[i];
  }
}

// ---------------------------------------------------------------------------
extern "C" void kernel_launch(void* const* d_in, const int* in_sizes, int n_in,
                              void* d_out, int out_size, void* d_ws, size_t ws_size,
                              hipStream_t stream) {
  (void)in_sizes; (void)n_in; (void)out_size; (void)ws_size;
  const float* x   = (const float*)d_in[0];
  const float* wu  = (const float*)d_in[1];
  const float* wf  = (const float*)d_in[2];
  const float* wo  = (const float*)d_in[3];
  const float* wc  = (const float*)d_in[4];
  const float* bu  = (const float*)d_in[5];
  const float* bfv = (const float*)d_in[6];
  const float* bo  = (const float*)d_in[7];
  const float* bc  = (const float*)d_in[8];
  const float* gwu = (const float*)d_in[9];
  const float* gwf = (const float*)d_in[10];
  const float* gwo = (const float*)d_in[11];
  const float* gwc = (const float*)d_in[12];
  const float* gbu = (const float*)d_in[13];
  const float* gbf = (const float*)d_in[14];
  const float* gbo = (const float*)d_in[15];
  const float* gbc = (const float*)d_in[16];

  // workspace: bf16 transposed weights (4 MB cell + 8 MB grid)
  unsigned short* cellW = (unsigned short*)d_ws;
  unsigned short* gridW = cellW + (size_t)4 * ND * NDIM * KC;

  pack_weights<<<1024, 256, 0, stream>>>(wu, wf, wo, wc, gwu, gwf, gwo, gwc, cellW, gridW);

  const size_t smem = (size_t)(ND * ROWS * NDIM + ROWS * KG) * sizeof(unsigned short)
                    + (size_t)2 * ND * ROWS * NDIM * sizeof(float);  // 192 KB < 320 KB/WGP
  hipFuncSetAttribute((const void*)grid_lstm_persistent,
                      hipFuncAttributeMaxDynamicSharedMemorySize, (int)smem);
  grid_lstm_persistent<<<NB / ROWS, WG, smem, stream>>>(
      x, bu, bfv, bo, bc, gbu, gbf, gbo, gbc, cellW, gridW, (float*)d_out);
}
